// new_layer_1262720385227
// MI455X (gfx1250) — matrix operand order repaired and checked by reading, not compiled, not run on hardware
//
#include <hip/hip_runtime.h>

// Problem constants (from reference)
#define B_     32
#define S_     512
#define H_     768
#define NH_    12
#define DH_    64
#define INTER_ 3072
#define E_     8

typedef __attribute__((ext_vector_type(16))) __bf16 v16bf;
typedef __attribute__((ext_vector_type(8)))  float  v8f;
typedef unsigned int u32x4 __attribute__((ext_vector_type(4)));
typedef int          i32x8 __attribute__((ext_vector_type(8)));
typedef int          i32x4 __attribute__((ext_vector_type(4)));

// ---------- bf16 helpers (round-to-nearest-even) ----------
__device__ __forceinline__ unsigned short f2bfu(float f) {
  union { float f; unsigned u; } x; x.f = f;
  unsigned r = x.u + 0x7FFFu + ((x.u >> 16) & 1u);
  return (unsigned short)(r >> 16);
}
__device__ __forceinline__ __bf16 u2bf(unsigned short u) {
  return __builtin_bit_cast(__bf16, u);
}

__device__ __forceinline__ v8f wmma_bf16(v16bf a, v16bf b, v8f c) {
  return __builtin_amdgcn_wmma_f32_16x16x32_bf16(false, a, false, b, (short)0, c,
                                                 false, false);
}

// A-matrix 16x32 bf16 fragment (ISA 7.12.2): lane l<16 row=l, K pairs
// {0..7,16..23}; lanes 16..31 same rows, K offset +8. element e -> K = hi*8+e+(e&8)
__device__ __forceinline__ v16bf fragA_lds(const unsigned short* base, int stride,
                                           int row0, int koff, int lane) {
  int l = lane & 15, hi = lane >> 4;
  const unsigned short* p = base + (size_t)(row0 + l) * stride + koff + (hi << 3);
  v16bf out;
#pragma unroll
  for (int e = 0; e < 16; ++e) out[e] = u2bf(p[e + (e & 8)]);
  return out;
}
__device__ __forceinline__ v16bf fragA_g(const unsigned short* base, int stride,
                                         int row0, int koff, int lane) {
  int l = lane & 15, hi = lane >> 4;
  const unsigned short* p = base + (size_t)(row0 + l) * stride + koff + (hi << 3);
  v16bf out;
#pragma unroll
  for (int e = 0; e < 16; ++e) out[e] = u2bf(p[e + (e & 8)]);
  return out;
}
// B-matrix 32x16 bf16 fragment: same K order as A (element e -> K = hi*8+e+(e&8)).
// Source stored transposed as [n][k] rows.
__device__ __forceinline__ v16bf fragB_lds(const unsigned short* base, int stride,
                                           int col0, int koff, int lane) {
  int l = lane & 15, hi = lane >> 4;
  const unsigned short* p = base + (size_t)(col0 + l) * stride + koff + (hi << 3);
  v16bf out;
#pragma unroll
  for (int e = 0; e < 16; ++e) out[e] = u2bf(p[e + (e & 8)]);
  return out;
}
__device__ __forceinline__ v16bf fragB_g_row(const unsigned short* base, int stride,
                                             int col0, int koff, int lane) {
  int l = lane & 15, hi = lane >> 4;
  const unsigned short* p = base + (size_t)(col0 + l) * stride + koff + (hi << 3);
  v16bf out;
#pragma unroll
  for (int e = 0; e < 16; ++e) out[e] = u2bf(p[e + (e & 8)]);
  return out;
}

__device__ __forceinline__ float gelu_tanh(float x) {
  float x3 = x * x * x;
  return 0.5f * x * (1.0f + tanhf(0.7978845608028654f * (x + 0.044715f * x3)));
}

// ---------- TDM: 2D bf16 tile -> LDS, with HW row padding ----------
// pad_enable: after every 16 DWORDs (32 bf16 = one tile row) insert 1 DWORD
// (2 bf16) -> LDS row stride 34 shorts (bank-skew), produced by the TDM.
// 6-arg builtin form (amdgpu-toolchain clang-23).
__device__ __forceinline__ void tdm_load_tile_bf16(
    unsigned lds_byte_off, const unsigned short* gptr,
    unsigned tile_k, unsigned tile_rows,         // tile dims (elements)
    unsigned tensor_k, unsigned tensor_rows,     // tensor dims (OOB bounds)
    unsigned stride_elems) {                     // dim0 stride (elements)
  unsigned long long ga = (unsigned long long)(size_t)gptr;
  u32x4 g0;
  g0[0] = 1u;                                        // count=1, user mode
  g0[1] = lds_byte_off;                              // lds_addr
  g0[2] = (unsigned)(ga & 0xffffffffu);              // global_addr[31:0]
  g0[3] = (unsigned)((ga >> 32) & 0x01ffffffu) | (2u << 30);  // [56:32] | type=2
  i32x8 g1;
  g1[0] = (int)((1u << 16) |                         // data_size = 2 bytes
                (1u << 20) |                         // pad_enable
                (3u << 22));                         // pad_interval: 16 DW, amount: 1 DW
  g1[1] = (int)((tensor_k & 0xffffu) << 16);         // tensor_dim0[15:0]
  g1[2] = (int)((tensor_k >> 16) | ((tensor_rows & 0xffffu) << 16));
  g1[3] = (int)((tensor_rows >> 16) | (tile_k << 16));  // tile_dim0
  g1[4] = (int)(tile_rows & 0xffffu);                // tile_dim1, tile_dim2=0
  g1[5] = (int)stride_elems;                         // tensor_dim0_stride[31:0]
  g1[6] = 0;
  g1[7] = 0;
  i32x4 g2 = {0, 0, 0, 0};
  i32x4 g3 = {0, 0, 0, 0};
  i32x8 g4 = {0, 0, 0, 0, 0, 0, 0, 0};
  __builtin_amdgcn_tensor_load_to_lds(g0, g1, g2, g3, g4, 0);
}

// ---------------- f32 -> bf16 flat convert ----------------
__global__ __launch_bounds__(256) void convert_bf16_kernel(
    const float* __restrict__ x, unsigned short* __restrict__ y, size_t n) {
  size_t i = (size_t)blockIdx.x * 256 + threadIdx.x;
  size_t step = (size_t)gridDim.x * 256;
  for (; i < n; i += step) y[i] = f2bfu(x[i]);
}

// ---------------- W[e][k][n] f32 -> Wt[e][n][k] bf16 ----------------
__global__ __launch_bounds__(256) void transpose_bf16_kernel(
    const float* __restrict__ W, unsigned short* __restrict__ Wt, int K, int N) {
  __shared__ float t[32][33];
  int k0 = blockIdx.x * 32, n0 = blockIdx.y * 32, e = blockIdx.z;
  const float* src = W + (size_t)e * K * N;
  unsigned short* dst = Wt + (size_t)e * N * K;
  int tx = threadIdx.x & 31, ty = threadIdx.x >> 5;  // 32 x 8
#pragma unroll
  for (int i = 0; i < 32; i += 8)
    t[ty + i][tx] = src[(size_t)(k0 + ty + i) * N + n0 + tx];
  __syncthreads();
#pragma unroll
  for (int i = 0; i < 32; i += 8)
    dst[(size_t)(n0 + ty + i) * K + k0 + tx] = f2bfu(t[tx][ty + i]);
}

// ---------------- Routing: sentence means ----------------
__global__ __launch_bounds__(256) void sent_mean_kernel(
    const float* __restrict__ hs, float* __restrict__ sent) {
  int b = blockIdx.x;
  for (int h = threadIdx.x; h < H_; h += 256) {
    float s = 0.f;
    for (int ss = 0; ss < S_; ++ss) s += hs[((size_t)b * S_ + ss) * H_ + h];
    sent[(size_t)b * H_ + h] = s * (1.0f / S_);
  }
}

// ---------------- Routing: center + init centers ----------------
__global__ __launch_bounds__(256) void center_init_kernel(
    const float* __restrict__ sent, float* __restrict__ latent,
    float* __restrict__ centers) {
  int h = blockIdx.x * 256 + threadIdx.x;
  if (h >= H_) return;
  float m = 0.f;
  for (int b = 0; b < B_; ++b) m += sent[(size_t)b * H_ + h];
  m *= (1.0f / B_);
  for (int b = 0; b < B_; ++b) {
    float c = sent[(size_t)b * H_ + h] - m;
    latent[(size_t)b * H_ + h] = c;
    if (b < E_) centers[(size_t)b * H_ + h] = c;
  }
}

// ---------------- Routing: k-means, single workgroup ----------------
__global__ __launch_bounds__(256) void kmeans_kernel(
    const float* __restrict__ latent, float* __restrict__ centers,
    int* __restrict__ rass) {
  __shared__ int s_assign[B_];
  __shared__ float s_cnt[E_];
  int tid = threadIdx.x;
  int b = tid >> 3, e = tid & 7;  // 32 x 8 = 256 (b,e) pairs
  for (int it = 0; it <= 20; ++it) {
    float d = 0.f;
    for (int h = 0; h < H_; ++h) {
      float x = latent[(size_t)b * H_ + h] - centers[(size_t)e * H_ + h];
      d += x * x;
    }
    int be = e;
    for (int o = 1; o < 8; o <<= 1) {
      float od = __shfl_xor(d, o, 8);
      int oe = __shfl_xor(be, o, 8);
      if (od < d || (od == d && oe < be)) { d = od; be = oe; }
    }
    if (e == 0) s_assign[b] = be;
    __syncthreads();
    if (it == 20) break;
    if (tid < E_) {
      float c = 0.f;
      for (int bb = 0; bb < B_; ++bb) c += (s_assign[bb] == tid) ? 1.f : 0.f;
      s_cnt[tid] = c;
    }
    __syncthreads();
    for (int h = tid; h < H_; h += 256) {
      float sums[E_];
#pragma unroll
      for (int ee = 0; ee < E_; ++ee) sums[ee] = 0.f;
      for (int bb = 0; bb < B_; ++bb)
        sums[s_assign[bb]] += latent[(size_t)bb * H_ + h];
#pragma unroll
      for (int ee = 0; ee < E_; ++ee)
        if (s_cnt[ee] > 0.f)
          centers[(size_t)ee * H_ + h] = sums[ee] / fmaxf(s_cnt[ee], 1.f);
    }
    __syncthreads();
  }
  if (tid < B_) rass[tid] = s_assign[tid];
}

// ---------------- Expert-gathered GEMM: Y = A @ Wt[r[b]]^T + bias -------------
// A: bf16 [B][M][K]; Wt: bf16 [E][N][K] (pre-transposed). Block tile 256x64,
// K-chunks of 32 staged by the Tensor Data Mover into double-buffered LDS.
// Each of 8 waves owns 2 M-frags x 4 N-frags = 8 WMMAs/chunk (B frags reused).
// out_mode: 0 = f32 [b][m][n], 1 = bf16 head-split (B,NH,S,DH),
//           2 = bf16 [b][m][n], 3 = bf16 transposed head-split (B,NH,DH,S)
__global__ __launch_bounds__(256) void gemm_expert(
    const unsigned short* __restrict__ A, const unsigned short* __restrict__ Wt,
    const float* __restrict__ bb, const int* __restrict__ rr,
    void* __restrict__ Y, int M, int N, int K, int do_gelu, int out_mode) {
  __shared__ unsigned short As[2][256][34];  // [m][k] bf16, stride 34 via TDM pad
  __shared__ unsigned short Bs[2][64][34];   // [n][k] bf16
  const int b = blockIdx.z;
  const int m0 = blockIdx.x * 256;
  const int n0 = blockIdx.y * 64;
  const int tid = threadIdx.x;
  const int lane = tid & 31;
  const int w = tid >> 5;  // 8 waves; wave w owns M-frags {2w,2w+1}, N-frags 0..3
  const int e = rr[b];
  const unsigned short* Ab = A + (size_t)b * M * K + (size_t)m0 * K;
  const unsigned short* Wm = Wt + (size_t)e * N * K + (size_t)n0 * K;
  const float* bias = bb + (size_t)e * N;

  const unsigned ldsA[2] = {(unsigned)(size_t)&As[0][0][0],
                            (unsigned)(size_t)&As[1][0][0]};
  const unsigned ldsB[2] = {(unsigned)(size_t)&Bs[0][0][0],
                            (unsigned)(size_t)&Bs[1][0][0]};
  const int KC = K >> 5;

  if (w == 0) {  // TDM issue: one instruction per wave, EXEC-independent
    tdm_load_tile_bf16(ldsA[0], Ab, 32, 256, (unsigned)K, 256, (unsigned)K);
    tdm_load_tile_bf16(ldsB[0], Wm, 32, 64, (unsigned)K, 64, (unsigned)K);
  }

  v8f acc0[4] = {v8f{}, v8f{}, v8f{}, v8f{}};
  v8f acc1[4] = {v8f{}, v8f{}, v8f{}, v8f{}};
  for (int kc = 0; kc < KC; ++kc) {
    const int buf = kc & 1;
    if (w == 0) __builtin_amdgcn_s_wait_tensorcnt(0);
    __syncthreads();  // publish TDM writes; fences prior LDS reads of this buf
    if (w == 0 && kc + 1 < KC) {
      const int nb = buf ^ 1;
      tdm_load_tile_bf16(ldsA[nb], Ab + ((kc + 1) << 5), 32, 256,
                         (unsigned)K, 256, (unsigned)K);
      tdm_load_tile_bf16(ldsB[nb], Wm + ((kc + 1) << 5), 32, 64,
                         (unsigned)K, 64, (unsigned)K);
    }
    // Hoist all fragment loads so DS waits overlap the WMMA stream.
    v16bf a0 = fragA_lds(&As[buf][0][0], 34, w * 32, 0, lane);
    v16bf a1 = fragA_lds(&As[buf][0][0], 34, w * 32 + 16, 0, lane);
    v16bf b0 = fragB_lds(&Bs[buf][0][0], 34, 0, 0, lane);
    v16bf b1 = fragB_lds(&Bs[buf][0][0], 34, 16, 0, lane);
    v16bf b2 = fragB_lds(&Bs[buf][0][0], 34, 32, 0, lane);
    v16bf b3 = fragB_lds(&Bs[buf][0][0], 34, 48, 0, lane);
    acc0[0] = wmma_bf16(a0, b0, acc0[0]);
    acc1[0] = wmma_bf16(a1, b0, acc1[0]);
    acc0[1] = wmma_bf16(a0, b1, acc0[1]);
    acc1[1] = wmma_bf16(a1, b1, acc1[1]);
    acc0[2] = wmma_bf16(a0, b2, acc0[2]);
    acc1[2] = wmma_bf16(a1, b2, acc1[2]);
    acc0[3] = wmma_bf16(a0, b3, acc0[3]);
    acc1[3] = wmma_bf16(a1, b3, acc1[3]);
  }
  // epilogue: C layout - element j -> (M = hi*8 + j, N = lane&15)
  const int l = lane & 15, hi = lane >> 4;
#pragma unroll
  for (int nf = 0; nf < 4; ++nf) {
    int n = n0 + nf * 16 + l;
    float bv = bias[n];
    int head = n >> 6, d = n & 63;
#pragma unroll
    for (int p = 0; p < 2; ++p) {
#pragma unroll
      for (int j = 0; j < 8; ++j) {
        int m = m0 + w * 32 + p * 16 + hi * 8 + j;
        float val = (p == 0 ? acc0[nf][j] : acc1[nf][j]) + bv;
        if (do_gelu) val = gelu_tanh(val);
        if (out_mode == 1) {
          ((unsigned short*)Y)[(((size_t)b * NH_ + head) * M + m) * DH_ + d] =
              f2bfu(val);
        } else if (out_mode == 2) {
          ((unsigned short*)Y)[(size_t)b * M * N + (size_t)m * N + n] = f2bfu(val);
        } else if (out_mode == 3) {
          ((unsigned short*)Y)[(((size_t)b * NH_ + head) * DH_ + d) * M + m] =
              f2bfu(val);
        } else {
          ((float*)Y)[(size_t)b * M * N + (size_t)m * N + n] = val;
        }
      }
    }
  }
}

// ---------------- Attention: per (b,h) 16-query tile (bf16 in/out) ------------
// q,k: (B,NH,S,DH) bf16; vt: (B,NH,DH,S) bf16 (transposed V for contiguous
// B-fragments in probs @ V).
__global__ __launch_bounds__(128) void attention_kernel(
    const unsigned short* __restrict__ q, const unsigned short* __restrict__ k,
    const unsigned short* __restrict__ vt, const float* __restrict__ mask,
    unsigned short* __restrict__ ctx) {
  __shared__ float s_sc[16][513];
  __shared__ unsigned short s_pr[16][528];
  const int bh = blockIdx.y;
  const int b = bh / NH_, h = bh % NH_;
  const int s0 = blockIdx.x * 16;
  const int tid = threadIdx.x, lane = tid & 31, w = tid >> 5;  // 4 waves
  const int l = lane & 15, hi = lane >> 4;
  const unsigned short* qb = q + (size_t)bh * S_ * DH_;
  const unsigned short* kb = k + (size_t)bh * S_ * DH_;
  const unsigned short* vtb = vt + (size_t)bh * DH_ * S_;

  // Phase 1: scores = Q K^T / 8 + mask penalty
  v16bf a0 = fragA_g(qb, DH_, s0, 0, lane);
  v16bf a1 = fragA_g(qb, DH_, s0, 32, lane);
  for (int t = w; t < 32; t += 4) {
    v8f acc = {};
    v16bf bk0 = fragB_g_row(kb, DH_, t * 16, 0, lane);
    v16bf bk1 = fragB_g_row(kb, DH_, t * 16, 32, lane);
    acc = wmma_bf16(a0, bk0, acc);
    acc = wmma_bf16(a1, bk1, acc);
    int col = t * 16 + l;
    float pen = (1.0f - mask[(size_t)b * S_ + col]) * -10000.0f;
#pragma unroll
    for (int j = 0; j < 8; ++j)
      s_sc[hi * 8 + j][col] = acc[j] * 0.125f + pen;
  }
  __syncthreads();
  // Phase 2: softmax rows (8 threads/row, shuffle reductions in wave32)
  {
    int row = tid >> 3, c0 = tid & 7;
    float m = -3.0e38f;
    for (int c = c0; c < S_; c += 8) m = fmaxf(m, s_sc[row][c]);
    for (int o = 1; o < 8; o <<= 1) m = fmaxf(m, __shfl_xor(m, o, 8));
    float sum = 0.f;
    for (int c = c0; c < S_; c += 8) {
      float ex = __expf(s_sc[row][c] - m);
      s_sc[row][c] = ex;
      sum += ex;
    }
    for (int o = 1; o < 8; o <<= 1) sum += __shfl_xor(sum, o, 8);
    float inv = 1.0f / sum;
    for (int c = c0; c < S_; c += 8) s_pr[row][c] = f2bfu(s_sc[row][c] * inv);
  }
  __syncthreads();
  // Phase 3: ctx = probs @ V ; wave w owns dh columns [w*16, w*16+16).
  // V^T rows are contiguous in keys -> plain row-major B fragments.
  v8f acc = {};
  for (int kc = 0; kc < 16; ++kc) {
    v16bf a = fragA_lds(&s_pr[0][0], 528, 0, kc * 32, lane);
    v16bf bv = fragB_g_row(vtb, S_, w * 16, kc * 32, lane);
    acc = wmma_bf16(a, bv, acc);
  }
#pragma unroll
  for (int j = 0; j < 8; ++j) {
    int srow = s0 + hi * 8 + j;
    ctx[((size_t)b * S_ + srow) * H_ + h * DH_ + w * 16 + l] = f2bfu(acc[j]);
  }
}

// ---------------- Fused residual + LayerNorm (f32 out + optional bf16 mirror) --
__global__ __launch_bounds__(256) void residual_ln_kernel(
    const float* __restrict__ xa, const float* __restrict__ xb,
    const float* __restrict__ g, const float* __restrict__ bt,
    const int* __restrict__ rr, float* __restrict__ y,
    unsigned short* __restrict__ ybf, int write_bf) {
  __shared__ float sh1[8], sh2[8];
  const int row = blockIdx.x;  // B*S rows
  const int b = row / S_;
  const int e = rr[b];
  const int tid = threadIdx.x;
  float vals[3];
  float s1 = 0.f, s2 = 0.f;
#pragma unroll
  for (int i = 0; i < 3; ++i) {
    int h = tid + i * 256;
    float v = xa[(size_t)row * H_ + h] + xb[(size_t)row * H_ + h];
    vals[i] = v;
    s1 += v;
    s2 += v * v;
  }
  for (int o = 16; o > 0; o >>= 1) {
    s1 += __shfl_xor(s1, o, 32);
    s2 += __shfl_xor(s2, o, 32);
  }
  if ((tid & 31) == 0) { sh1[tid >> 5] = s1; sh2[tid >> 5] = s2; }
  __syncthreads();
  float t1 = 0.f, t2 = 0.f;
#pragma unroll
  for (int i = 0; i < 8; ++i) { t1 += sh1[i]; t2 += sh2[i]; }
  float mu = t1 * (1.0f / H_);
  float var = t2 * (1.0f / H_) - mu * mu;
  float rs = rsqrtf(var + 1e-12f);
#pragma unroll
  for (int i = 0; i < 3; ++i) {
    int h = tid + i * 256;
    float o = (vals[i] - mu) * rs * g[(size_t)e * H_ + h] + bt[(size_t)e * H_ + h];
    y[(size_t)row * H_ + h] = o;
    if (write_bf) ybf[(size_t)row * H_ + h] = f2bfu(o);
  }
}

// ---------------- Host-side orchestration ----------------
extern "C" void kernel_launch(void* const* d_in, const int* in_sizes, int n_in,
                              void* d_out, int out_size, void* d_ws, size_t ws_size,
                              hipStream_t stream) {
  (void)in_sizes; (void)n_in; (void)out_size; (void)ws_size;
  const float* hs   = (const float*)d_in[0];
  const float* mask = (const float*)d_in[1];
  const float* Wq = (const float*)d_in[2];  const float* bq = (const float*)d_in[3];
  const float* Wk = (const float*)d_in[4];  const float* bk = (const float*)d_in[5];
  const float* Wv = (const float*)d_in[6];  const float* bv = (const float*)d_in[7];
  const float* Wo = (const float*)d_in[8];  const float* bo = (const float*)d_in[9];
  const float* g1 = (const float*)d_in[10]; const float* be1 = (const float*)d_in[11];
  const float* W1 = (const float*)d_in[12]; const float* b1  = (const float*)d_in[13];
  const float* W2 = (const float*)d_in[14]; const float* b2  = (const float*)d_in[15];
  const float* g2 = (const float*)d_in[16]; const float* be2 = (const float*)d_in[17];
  float* out = (float*)d_out;

  char* ws = (char*)d_ws;
  size_t off = 0;
  auto allocf = [&](size_t n) {
    float* p = (float*)(ws + off);
    off = (off + n * 4 + 255) & ~(size_t)255;
    return p;
  };
  auto allocb = [&](size_t n) {
    unsigned short* p = (unsigned short*)(ws + off);
    off = (off + n * 2 + 255) & ~(size_t)255;
    return p;
  };
  const size_t BSH = (size_t)B_ * S_ * H_;
  float* sent    = allocf((size_t)B_ * H_);
  float* latent  = allocf((size_t)B_ * H_);
  float* centers = allocf((size_t)E_ * H_);
  int*   rass    = (int*)allocf(B_);
  unsigned short* hs_bf = allocb(BSH);
  unsigned short* Wtq = allocb((size_t)E_ * H_ * H_);
  unsigned short* Wtk = allocb((size_t)E_ * H_ * H_);
  unsigned short* Wtv = allocb((size_t)E_ * H_ * H_);
  unsigned short* Wto = allocb((size_t)E_ * H_ * H_);
  unsigned short* Wt1 = allocb((size_t)E_ * H_ * INTER_);
  unsigned short* Wt2 = allocb((size_t)E_ * INTER_ * H_);
  unsigned short* qb  = allocb(BSH);
  unsigned short* kb  = allocb(BSH);
  unsigned short* vtb = allocb(BSH);
  unsigned short* ctx_bf = allocb(BSH);
  unsigned short* att_bf = allocb(BSH);
  unsigned short* h1_bf  = allocb((size_t)B_ * S_ * INTER_);
  float* attp = allocf(BSH);
  float* att  = allocf(BSH);
  float* ffn  = allocf(BSH);

  // One-time precision/layout conversion (bandwidth-amortized)
  convert_bf16_kernel<<<2048, 256, 0, stream>>>(hs, hs_bf, BSH);
  dim3 tHH(H_ / 32, H_ / 32, E_), tHI(H_ / 32, INTER_ / 32, E_),
       tIH(INTER_ / 32, H_ / 32, E_);
  transpose_bf16_kernel<<<tHH, 256, 0, stream>>>(Wq, Wtq, H_, H_);
  transpose_bf16_kernel<<<tHH, 256, 0, stream>>>(Wk, Wtk, H_, H_);
  transpose_bf16_kernel<<<tHH, 256, 0, stream>>>(Wv, Wtv, H_, H_);
  transpose_bf16_kernel<<<tHH, 256, 0, stream>>>(Wo, Wto, H_, H_);
  transpose_bf16_kernel<<<tHI, 256, 0, stream>>>(W1, Wt1, H_, INTER_);
  transpose_bf16_kernel<<<tIH, 256, 0, stream>>>(W2, Wt2, INTER_, H_);

  // Routing
  sent_mean_kernel<<<B_, 256, 0, stream>>>(hs, sent);
  center_init_kernel<<<(H_ + 255) / 256, 256, 0, stream>>>(sent, latent, centers);
  kmeans_kernel<<<1, 256, 0, stream>>>(latent, centers, rass);

  // QKV projections (bf16 head-split writeout; V transposed per head)
  dim3 g768(S_ / 256, H_ / 64, B_);
  gemm_expert<<<g768, 256, 0, stream>>>(hs_bf, Wtq, bq, rass, qb, S_, H_, H_, 0, 1);
  gemm_expert<<<g768, 256, 0, stream>>>(hs_bf, Wtk, bk, rass, kb, S_, H_, H_, 0, 1);
  gemm_expert<<<g768, 256, 0, stream>>>(hs_bf, Wtv, bv, rass, vtb, S_, H_, H_, 0, 3);

  // Attention (bf16 ctx out)
  attention_kernel<<<dim3(S_ / 16, B_ * NH_), 128, 0, stream>>>(qb, kb, vtb, mask,
                                                                ctx_bf);

  // Output projection + residual LN1 (f32 + bf16 mirror for FFN input)
  gemm_expert<<<g768, 256, 0, stream>>>(ctx_bf, Wto, bo, rass, attp, S_, H_, H_,
                                        0, 0);
  residual_ln_kernel<<<B_ * S_, 256, 0, stream>>>(attp, hs, g1, be1, rass, att,
                                                  att_bf, 1);

  // FFN + residual LN2 -> out
  gemm_expert<<<dim3(S_ / 256, INTER_ / 64, B_), 256, 0, stream>>>(
      att_bf, Wt1, b1, rass, h1_bf, S_, INTER_, H_, 1, 2);
  gemm_expert<<<g768, 256, 0, stream>>>(h1_bf, Wt2, b2, rass, ffn, S_, H_,
                                        INTER_, 0, 0);
  residual_ln_kernel<<<B_ * S_, 256, 0, stream>>>(att, ffn, g2, be2, rass, out,
                                                  att_bf, 0);
}
